// CausalSelfAttention_33560874451600
// MI455X (gfx1250) — matrix operand order repaired and checked by reading, not compiled, not run on hardware
//
#include <hip/hip_runtime.h>
#include <hip/hip_bf16.h>

// ---------------------------------------------------------------------------
// CausalSelfAttention forward for MI455X (gfx1250, wave32, WMMA).
// v2: register-blocked GEMMs (32x64 tile / wave, 8 accumulators) and
// 32-query-row flash-attention waves (K/V loads amortized across 2 q-tiles).
// ---------------------------------------------------------------------------

typedef __attribute__((ext_vector_type(16))) _Float16 v16h;
typedef __attribute__((ext_vector_type(8)))  _Float16 v8h;
typedef __attribute__((ext_vector_type(8)))  float    v8f;

constexpr int EMB  = 1024;
constexpr int NH   = 16;
constexpr int HD   = 64;      // head dim
constexpr int SQ   = 2048;    // sequence length
constexpr int BB   = 4;       // batch
constexpr int MTOT = BB * SQ; // 8192 rows

// ---- WMMA helper -----------------------------------------------------------
__device__ __forceinline__ v8f wmma_f16(v16h a, v16h b, v8f c) {
    return __builtin_amdgcn_wmma_f32_16x16x32_f16(
        false, a, false, b, (short)0, c, false, false);
}

// ---- A-tile loader: 16x32 f16, row-major source ----------------------------
__device__ __forceinline__ v16h load_a_tile(const _Float16* base, int ld,
                                            int row0, int k0, int lane) {
    int m = lane & 15, hi = lane >> 4;
    const _Float16* p = base + (size_t)(row0 + m) * ld + k0 + hi * 8;
    v8h lo = *(const v8h*)p;
    v8h hh = *(const v8h*)(p + 16);
    v16h a;
#pragma unroll
    for (int i = 0; i < 8; ++i) { a[i] = lo[i]; a[i + 8] = hh[i]; }
    return a;
}

// ---- B-tile loader: 32x16 f16 from K-major source --------------------------
__device__ __forceinline__ v16h load_b_tile(const _Float16* base, int ld,
                                            int k0, int n0, int lane) {
    int n = lane & 15, hi = lane >> 4;
    const _Float16* p = base + (size_t)(k0 + hi * 8) * ld + n0 + n;
    v16h b;
#pragma unroll
    for (int i = 0; i < 8; ++i) { b[i] = p[(size_t)i * ld]; b[i + 8] = p[(size_t)(16 + i) * ld]; }
    return b;
}

// ---- conversion kernels ----------------------------------------------------
__global__ __launch_bounds__(256) void cvt_f32_to_f16(
    const float* __restrict__ in, _Float16* __restrict__ out, int n) {
    int i = blockIdx.x * 256 + threadIdx.x;
    if (i < n) out[i] = (_Float16)in[i];
}

// Wt[k*1024 + n] = W[n*1024 + k]
__global__ __launch_bounds__(256) void cvt_transpose_f16(
    const float* __restrict__ W, _Float16* __restrict__ Wt) {
    int i = blockIdx.x * 256 + threadIdx.x;
    int n = i >> 10, k = i & 1023;
    Wt[(size_t)k * EMB + n] = (_Float16)W[i];
}

// ---- projection GEMM: Out(f16) = X @ Wt + bias, 32x64 tile per wave --------
// kTrans==0: store [B,H,S,D];  kTrans==1: store [B,H,D,S] (K matrix).
__global__ __launch_bounds__(256) void proj_gemm_f16out(
    const _Float16* __restrict__ X, const _Float16* __restrict__ Wt,
    const float* __restrict__ bias, _Float16* __restrict__ Out, int kTrans) {
    int lane = threadIdx.x & 31, widx = threadIdx.x >> 5;
    int waveId = blockIdx.x * 8 + widx;       // 4096 waves
    int tm = waveId >> 4, tn = waveId & 15;   // 256 x 16 macro-tiles
    int row0 = tm * 32, n0 = tn * 64;

    v8f acc[2][4];
#pragma unroll
    for (int i = 0; i < 2; ++i)
#pragma unroll
        for (int j = 0; j < 4; ++j) acc[i][j] = (v8f){};

    for (int k0 = 0; k0 < EMB; k0 += 32) {
        if (k0 + 32 < EMB) {
            __builtin_prefetch(X + (size_t)(row0 + (lane & 15)) * EMB + k0 + 32, 0, 1);
            __builtin_prefetch(Wt + (size_t)(k0 + 32 + lane) * EMB + n0, 0, 1);
        }
        v16h a0 = load_a_tile(X, EMB, row0,      k0, lane);
        v16h a1 = load_a_tile(X, EMB, row0 + 16, k0, lane);
        v16h b0 = load_b_tile(Wt, EMB, k0, n0,      lane);
        v16h b1 = load_b_tile(Wt, EMB, k0, n0 + 16, lane);
        v16h b2 = load_b_tile(Wt, EMB, k0, n0 + 32, lane);
        v16h b3 = load_b_tile(Wt, EMB, k0, n0 + 48, lane);
        acc[0][0] = wmma_f16(a0, b0, acc[0][0]);
        acc[0][1] = wmma_f16(a0, b1, acc[0][1]);
        acc[0][2] = wmma_f16(a0, b2, acc[0][2]);
        acc[0][3] = wmma_f16(a0, b3, acc[0][3]);
        acc[1][0] = wmma_f16(a1, b0, acc[1][0]);
        acc[1][1] = wmma_f16(a1, b1, acc[1][1]);
        acc[1][2] = wmma_f16(a1, b2, acc[1][2]);
        acc[1][3] = wmma_f16(a1, b3, acc[1][3]);
    }

    int ln = lane & 15, hi = lane >> 4;
#pragma unroll
    for (int j = 0; j < 4; ++j) {
        int nn = n0 + j * 16 + ln;
        float bvv = bias[nn];
        int h = nn >> 6, d = nn & 63;
#pragma unroll
        for (int i = 0; i < 2; ++i) {
#pragma unroll
            for (int r = 0; r < 8; ++r) {
                int mrow = row0 + i * 16 + r + 8 * hi;
                int b_ = mrow >> 11;
                int s_ = mrow & 2047;
                float v = acc[i][j][r] + bvv;
                size_t idx = kTrans
                    ? (((size_t)(b_ * NH + h) * HD + d) * SQ + s_)
                    : (((size_t)(b_ * NH + h) * SQ + s_) * HD + d);
                Out[idx] = (_Float16)v;
            }
        }
    }
}

// ---- output GEMM: out(f32) = Ctx @ WoT + bo, 32x64 tile per wave -----------
__global__ __launch_bounds__(256) void out_gemm_f32(
    const _Float16* __restrict__ X, const _Float16* __restrict__ Wt,
    const float* __restrict__ bias, float* __restrict__ Out) {
    int lane = threadIdx.x & 31, widx = threadIdx.x >> 5;
    int waveId = blockIdx.x * 8 + widx;
    int tm = waveId >> 4, tn = waveId & 15;
    int row0 = tm * 32, n0 = tn * 64;

    v8f acc[2][4];
#pragma unroll
    for (int i = 0; i < 2; ++i)
#pragma unroll
        for (int j = 0; j < 4; ++j) acc[i][j] = (v8f){};

    for (int k0 = 0; k0 < EMB; k0 += 32) {
        if (k0 + 32 < EMB) {
            __builtin_prefetch(X + (size_t)(row0 + (lane & 15)) * EMB + k0 + 32, 0, 1);
            __builtin_prefetch(Wt + (size_t)(k0 + 32 + lane) * EMB + n0, 0, 1);
        }
        v16h a0 = load_a_tile(X, EMB, row0,      k0, lane);
        v16h a1 = load_a_tile(X, EMB, row0 + 16, k0, lane);
        v16h b0 = load_b_tile(Wt, EMB, k0, n0,      lane);
        v16h b1 = load_b_tile(Wt, EMB, k0, n0 + 16, lane);
        v16h b2 = load_b_tile(Wt, EMB, k0, n0 + 32, lane);
        v16h b3 = load_b_tile(Wt, EMB, k0, n0 + 48, lane);
        acc[0][0] = wmma_f16(a0, b0, acc[0][0]);
        acc[0][1] = wmma_f16(a0, b1, acc[0][1]);
        acc[0][2] = wmma_f16(a0, b2, acc[0][2]);
        acc[0][3] = wmma_f16(a0, b3, acc[0][3]);
        acc[1][0] = wmma_f16(a1, b0, acc[1][0]);
        acc[1][1] = wmma_f16(a1, b1, acc[1][1]);
        acc[1][2] = wmma_f16(a1, b2, acc[1][2]);
        acc[1][3] = wmma_f16(a1, b3, acc[1][3]);
    }

    int ln = lane & 15, hi = lane >> 4;
#pragma unroll
    for (int j = 0; j < 4; ++j) {
        int nn = n0 + j * 16 + ln;
        float bvv = bias[nn];
#pragma unroll
        for (int i = 0; i < 2; ++i) {
#pragma unroll
            for (int r = 0; r < 8; ++r) {
                int mrow = row0 + i * 16 + r + 8 * hi;
                Out[(size_t)mrow * EMB + nn] = acc[i][j][r] + bvv;
            }
        }
    }
}

// ---- flash attention: one wave = 32 query rows (two 16-row tiles) ----------
// Q:[B,H,S,D]  Kt:[B,H,D,S]  V:[B,H,S,D]  Ctx:[B,S,E] f16
__global__ __launch_bounds__(256) void attn_kernel(
    const _Float16* __restrict__ Q, const _Float16* __restrict__ Kt,
    const _Float16* __restrict__ V, _Float16* __restrict__ Ctx) {
    __shared__ __align__(16) _Float16 pbuf[8][2][16 * 32];

    int lane = threadIdx.x & 31, widx = threadIdx.x >> 5;
    int waveId = blockIdx.x * 8 + widx;   // 4096 waves = 64 bh * 64 q-tiles
    int bh = waveId >> 6, qt = waveId & 63;
    int q0 = qt * 32;
    int ln = lane & 15, hi = lane >> 4;

    const _Float16* Qb = Q  + (size_t)bh * SQ * HD;
    const _Float16* Kb = Kt + (size_t)bh * HD * SQ;
    const _Float16* Vb = V  + (size_t)bh * SQ * HD;

    // 32x64 Q block: 2 row-tiles x 2 K32 chunks
    v16h aq[2][2];
#pragma unroll
    for (int i = 0; i < 2; ++i) {
        aq[i][0] = load_a_tile(Qb, HD, q0 + 16 * i, 0,  lane);
        aq[i][1] = load_a_tile(Qb, HD, q0 + 16 * i, 32, lane);
    }

    v8f o[2][4];
    float mrow[2][8], lrow[2][8];
#pragma unroll
    for (int i = 0; i < 2; ++i) {
#pragma unroll
        for (int j = 0; j < 4; ++j) o[i][j] = (v8f){};
#pragma unroll
        for (int r = 0; r < 8; ++r) { mrow[i][r] = -1e30f; lrow[i][r] = 0.f; }
    }

    const float scale = 0.125f;      // 1/sqrt(64)
    int jcmax = (q0 + 31) >> 5;      // chunks through the diagonal

    for (int jc = 0; jc <= jcmax; ++jc) {
        int j0 = jc * 32;
        // K B-tiles shared by both query row-tiles
        v16h bk0a = load_b_tile(Kb, SQ, 0,  j0,      lane);
        v16h bk0b = load_b_tile(Kb, SQ, 32, j0,      lane);
        v16h bk1a = load_b_tile(Kb, SQ, 0,  j0 + 16, lane);
        v16h bk1b = load_b_tile(Kb, SQ, 32, j0 + 16, lane);

        int kp0 = j0 + ln, kp1 = kp0 + 16;

#pragma unroll
        for (int i = 0; i < 2; ++i) {
            v8f s0 = {}, s1 = {};
            s0 = wmma_f16(aq[i][0], bk0a, s0);
            s0 = wmma_f16(aq[i][1], bk0b, s0);
            s1 = wmma_f16(aq[i][0], bk1a, s1);
            s1 = wmma_f16(aq[i][1], bk1b, s1);

            _Float16* pb = &pbuf[widx][i][0];
#pragma unroll
            for (int r = 0; r < 8; ++r) {
                int qr = q0 + 16 * i + r + 8 * hi;
                float v0 = (kp0 <= qr) ? s0[r] * scale : -1e30f;
                float v1 = (kp1 <= qr) ? s1[r] * scale : -1e30f;
                float rm = fmaxf(v0, v1);
                rm = fmaxf(rm, __shfl_xor(rm, 1, 16));
                rm = fmaxf(rm, __shfl_xor(rm, 2, 16));
                rm = fmaxf(rm, __shfl_xor(rm, 4, 16));
                rm = fmaxf(rm, __shfl_xor(rm, 8, 16));
                float mn = fmaxf(mrow[i][r], rm);
                float corr = __expf(mrow[i][r] - mn);
                mrow[i][r] = mn;
                float p0 = __expf(v0 - mn);
                float p1 = __expf(v1 - mn);
                float ps = p0 + p1;
                ps += __shfl_xor(ps, 1, 16);
                ps += __shfl_xor(ps, 2, 16);
                ps += __shfl_xor(ps, 4, 16);
                ps += __shfl_xor(ps, 8, 16);
                lrow[i][r] = lrow[i][r] * corr + ps;
                o[i][0][r] *= corr; o[i][1][r] *= corr;
                o[i][2][r] *= corr; o[i][3][r] *= corr;
                pb[(r + 8 * hi) * 32 + ln]      = (_Float16)p0;
                pb[(r + 8 * hi) * 32 + 16 + ln] = (_Float16)p1;
            }
        }
        asm volatile("s_wait_dscnt 0" ::: "memory");

        // V B-tiles shared by both query row-tiles
        v16h bv0 = load_b_tile(Vb, HD, j0, 0,  lane);
        v16h bv1 = load_b_tile(Vb, HD, j0, 16, lane);
        v16h bv2 = load_b_tile(Vb, HD, j0, 32, lane);
        v16h bv3 = load_b_tile(Vb, HD, j0, 48, lane);
#pragma unroll
        for (int i = 0; i < 2; ++i) {
            v16h ap = load_a_tile(&pbuf[widx][i][0], 32, 0, 0, lane);
            o[i][0] = wmma_f16(ap, bv0, o[i][0]);
            o[i][1] = wmma_f16(ap, bv1, o[i][1]);
            o[i][2] = wmma_f16(ap, bv2, o[i][2]);
            o[i][3] = wmma_f16(ap, bv3, o[i][3]);
        }
    }

    int b_ = bh >> 4, h_ = bh & 15;
#pragma unroll
    for (int i = 0; i < 2; ++i) {
#pragma unroll
        for (int r = 0; r < 8; ++r) {
            float inv = 1.0f / lrow[i][r];
            int s_ = q0 + 16 * i + r + 8 * hi;
            size_t base = ((size_t)b_ * SQ + s_) * EMB + h_ * HD;
            Ctx[base + 0  + ln] = (_Float16)(o[i][0][r] * inv);
            Ctx[base + 16 + ln] = (_Float16)(o[i][1][r] * inv);
            Ctx[base + 32 + ln] = (_Float16)(o[i][2][r] * inv);
            Ctx[base + 48 + ln] = (_Float16)(o[i][3][r] * inv);
        }
    }
}

// ---------------------------------------------------------------------------
extern "C" void kernel_launch(void* const* d_in, const int* in_sizes, int n_in,
                              void* d_out, int out_size, void* d_ws, size_t ws_size,
                              hipStream_t stream) {
    const float* x  = (const float*)d_in[0];
    const float* Wq = (const float*)d_in[1];
    const float* bq = (const float*)d_in[2];
    const float* Wk = (const float*)d_in[3];
    const float* bk = (const float*)d_in[4];
    const float* Wv = (const float*)d_in[5];
    const float* bv = (const float*)d_in[6];
    const float* Wo = (const float*)d_in[7];
    const float* bo = (const float*)d_in[8];
    float* out = (float*)d_out;

    _Float16* ws = (_Float16*)d_ws;
    size_t off = 0;
    _Float16* x16   = ws + off; off += (size_t)MTOT * EMB;
    _Float16* wqT   = ws + off; off += (size_t)EMB * EMB;
    _Float16* wkT   = ws + off; off += (size_t)EMB * EMB;
    _Float16* wvT   = ws + off; off += (size_t)EMB * EMB;
    _Float16* woT   = ws + off; off += (size_t)EMB * EMB;
    _Float16* q16   = ws + off; off += (size_t)MTOT * EMB;
    _Float16* kT16  = ws + off; off += (size_t)MTOT * EMB;
    _Float16* v16p  = ws + off; off += (size_t)MTOT * EMB;
    _Float16* ctx16 = ws + off; off += (size_t)MTOT * EMB;

    cvt_f32_to_f16<<<(MTOT * EMB) / 256, 256, 0, stream>>>(x, x16, MTOT * EMB);
    cvt_transpose_f16<<<(EMB * EMB) / 256, 256, 0, stream>>>(Wq, wqT);
    cvt_transpose_f16<<<(EMB * EMB) / 256, 256, 0, stream>>>(Wk, wkT);
    cvt_transpose_f16<<<(EMB * EMB) / 256, 256, 0, stream>>>(Wv, wvT);
    cvt_transpose_f16<<<(EMB * EMB) / 256, 256, 0, stream>>>(Wo, woT);

    proj_gemm_f16out<<<512, 256, 0, stream>>>(x16, wqT, bq, q16, 0);
    proj_gemm_f16out<<<512, 256, 0, stream>>>(x16, wkT, bk, kT16, 1);
    proj_gemm_f16out<<<512, 256, 0, stream>>>(x16, wvT, bv, v16p, 0);

    attn_kernel<<<512, 256, 0, stream>>>(q16, kT16, v16p, ctx16);

    out_gemm_f32<<<512, 256, 0, stream>>>(ctx16, woT, bo, out);
}
